// GCN_12386685681966
// MI455X (gfx1250) — matrix operand order repaired and checked by reading, not compiled, not run on hardware
//
#include <hip/hip_runtime.h>
#include <hip/hip_bf16.h>

#define NNODES 50000
#define NEDGES 800000
#define IN_FEAT 512
#define HID 256
#define OUT_FEAT 64
#define BN_EPS 1e-5f

typedef __attribute__((ext_vector_type(16))) __bf16 v16bf;
typedef __attribute__((ext_vector_type(8)))  float  v8f;

// ---------------------------------------------------------------------------
// Pack a column-major-ish WMMA B fragment buffer from a row-major f32 weight
// matrix W[K][ncols].  Fragment layout (per 16x16x32 bf16 WMMA, wave32):
//   B element e of lane l  ->  K = kstep*32 + (l<16 ? 0 : 8) + (e<8 ? e : e+8),  N = colTile*16 + (l&15)
// Buffer layout: frag[(colTile*ksteps + kstep)][lane][16 bf16]  (32 B per lane).
// ---------------------------------------------------------------------------
__global__ void gcn_pack_w_bf16(const float* __restrict__ W, __bf16* __restrict__ P,
                                int K, int ncols) {
    int ksteps  = K >> 5;
    int colTile = blockIdx.x / ksteps;
    int kstep   = blockIdx.x - colTile * ksteps;
    int lane    = threadIdx.x;            // 0..31
    int n       = colTile * 16 + (lane & 15);
    int kbase   = kstep * 32 + ((lane < 16) ? 0 : 8);
    __bf16* dst = P + (size_t)blockIdx.x * 512 + (size_t)lane * 16;
#pragma unroll
    for (int e = 0; e < 16; ++e)
        dst[e] = (__bf16)W[(size_t)(kbase + (e < 8 ? e : e + 8)) * ncols + n];
}

// ---------------------------------------------------------------------------
// One wave computes a 16x64 tile of Out = A(f32,[rows x K]) * W(bf16 packed).
// A fragment layout (16-bit A 16x32): lane l holds row M = l&15;
//   element e -> K = kstep*32 + (l<16?0:8) + (e<8 ? e : e+8).
// C/D layout: VGPR r, lanes 0-15 -> (M=r, N=lane), lanes 16-31 -> (M=r+8, N=lane-16).
// ---------------------------------------------------------------------------
__device__ __forceinline__ void gcn_wmma_tile_16x64(
    const float* __restrict__ A, const __bf16* __restrict__ P,
    float* __restrict__ Out, int K, int ncols, int rowTile, int ct0) {
    const int lane  = threadIdx.x & 31;
    const int row   = rowTile * 16 + (lane & 15);
    const int koffA = (lane < 16) ? 0 : 8;
    const float* arow = A + (size_t)row * K;
    const int ksteps = K >> 5;
    const v16bf* __restrict__ PB = (const v16bf*)P;

    v8f c0 = {}, c1 = {}, c2 = {}, c3 = {};

    for (int ks = 0; ks < ksteps; ++ks) {
        const int kk = ks * 32 + koffA;
        float4 f0 = *(const float4*)(arow + kk);
        float4 f1 = *(const float4*)(arow + kk + 4);
        float4 f2 = *(const float4*)(arow + kk + 16);
        float4 f3 = *(const float4*)(arow + kk + 20);
        v16bf a;
        a[0]  = (__bf16)f0.x; a[1]  = (__bf16)f0.y; a[2]  = (__bf16)f0.z; a[3]  = (__bf16)f0.w;
        a[4]  = (__bf16)f1.x; a[5]  = (__bf16)f1.y; a[6]  = (__bf16)f1.z; a[7]  = (__bf16)f1.w;
        a[8]  = (__bf16)f2.x; a[9]  = (__bf16)f2.y; a[10] = (__bf16)f2.z; a[11] = (__bf16)f2.w;
        a[12] = (__bf16)f3.x; a[13] = (__bf16)f3.y; a[14] = (__bf16)f3.z; a[15] = (__bf16)f3.w;

        v16bf b0 = PB[(size_t)((ct0 + 0) * ksteps + ks) * 32 + lane];
        v16bf b1 = PB[(size_t)((ct0 + 1) * ksteps + ks) * 32 + lane];
        v16bf b2 = PB[(size_t)((ct0 + 2) * ksteps + ks) * 32 + lane];
        v16bf b3 = PB[(size_t)((ct0 + 3) * ksteps + ks) * 32 + lane];

        c0 = __builtin_amdgcn_wmma_f32_16x16x32_bf16(false, a, false, b0, (short)0, c0, false, false);
        c1 = __builtin_amdgcn_wmma_f32_16x16x32_bf16(false, a, false, b1, (short)0, c1, false, false);
        c2 = __builtin_amdgcn_wmma_f32_16x16x32_bf16(false, a, false, b2, (short)0, c2, false, false);
        c3 = __builtin_amdgcn_wmma_f32_16x16x32_bf16(false, a, false, b3, (short)0, c3, false, false);
    }

    const int m0   = (lane < 16) ? 0 : 8;
    const int ncol = lane & 15;
#pragma unroll
    for (int r = 0; r < 8; ++r) {
        size_t orow = (size_t)(rowTile * 16 + m0 + r) * ncols;
        Out[orow + (ct0 + 0) * 16 + ncol] = c0[r];
        Out[orow + (ct0 + 1) * 16 + ncol] = c1[r];
        Out[orow + (ct0 + 2) * 16 + ncol] = c2[r];
        Out[orow + (ct0 + 3) * 16 + ncol] = c3[r];
    }
}

// Layer 1: X[N,512] @ W1 -> XW[N,256].  4 waves/block cover cols 0..255 of one row tile.
__global__ void gcn_gemm1(const float* __restrict__ X, const __bf16* __restrict__ P,
                          float* __restrict__ Out) {
    int wave = threadIdx.x >> 5;
    gcn_wmma_tile_16x64(X, P, Out, IN_FEAT, HID, blockIdx.x, wave * 4);
}

// Layer 2: H[N,256] @ W2 -> H2[N,64].  Each wave owns one row tile, all 64 cols.
__global__ void gcn_gemm2(const float* __restrict__ H, const __bf16* __restrict__ P,
                          float* __restrict__ Out) {
    int wave = threadIdx.x >> 5;
    int rowTile = blockIdx.x * 4 + wave;
    if (rowTile >= NNODES / 16) return;   // wave-uniform exit (EXEC stays all-ones for WMMA)
    gcn_wmma_tile_16x64(H, P, Out, HID, OUT_FEAT, rowTile, 0);
}

// ---------------------------------------------------------------------------
__global__ void gcn_fill_zero4(float* __restrict__ p, long n4) {
    long i = (long)blockIdx.x * blockDim.x + threadIdx.x;
    if (i < n4) ((float4*)p)[i] = make_float4(0.f, 0.f, 0.f, 0.f);
}

// SpMM scatter:  out[rows[e], :] += vals[e] * in[cols[e], :]
// thread = one float4 (4 features) of one edge; hw fp32 atomics to L2-resident out.
__global__ void gcn_spmm_atomic(const int* __restrict__ rows, const int* __restrict__ cols,
                                const float* __restrict__ vals,
                                const float* __restrict__ in, float* __restrict__ out,
                                int F4, long total) {
    long tid = (long)blockIdx.x * blockDim.x + threadIdx.x;
    if (tid >= total) return;
    long e = tid / F4;
    int  q = (int)(tid - e * F4);
    int  r = rows[e];
    int  c = cols[e];
    float v = vals[e];
    float4 d = ((const float4*)in)[(size_t)c * F4 + q];
    float* o = out + (size_t)r * (F4 * 4) + (size_t)q * 4;
    unsafeAtomicAdd(o + 0, v * d.x);
    unsafeAtomicAdd(o + 1, v * d.y);
    unsafeAtomicAdd(o + 2, v * d.z);
    unsafeAtomicAdd(o + 3, v * d.w);
}

// Per-feature sum / sum-of-squares over all N rows.  blockDim.x == F (coalesced rows).
__global__ void gcn_col_stats(const float* __restrict__ in,
                              float* __restrict__ sums, float* __restrict__ sumsq, int F) {
    int f = threadIdx.x;
    float s = 0.f, s2 = 0.f;
    for (int r = blockIdx.x; r < NNODES; r += gridDim.x) {
        float v = in[(size_t)r * F + f];
        s += v;
        s2 += v * v;
    }
    unsafeAtomicAdd(&sums[f], s);
    unsafeAtomicAdd(&sumsq[f], s2);
}

// Fused BatchNorm (batch stats) + ReLU, in place.
__global__ void gcn_bn_relu(float* __restrict__ h,
                            const float* __restrict__ sums, const float* __restrict__ sumsq,
                            const float* __restrict__ gamma, const float* __restrict__ beta,
                            int F, long total) {
    long i = (long)blockIdx.x * blockDim.x + threadIdx.x;
    if (i >= total) return;
    int f = (int)(i & (F - 1));
    float mean = sums[f] * (1.f / NNODES);
    float var  = sumsq[f] * (1.f / NNODES) - mean * mean;
    float y = gamma[f] * (h[i] - mean) * rsqrtf(var + BN_EPS) + beta[f];
    h[i] = fmaxf(y, 0.f);
}

// Fused BatchNorm + log_softmax over 64 classes; one wave per row, 2 classes/lane.
__global__ void gcn_bn_logsoftmax(const float* __restrict__ in,
                                  const float* __restrict__ sums, const float* __restrict__ sumsq,
                                  const float* __restrict__ gamma, const float* __restrict__ beta,
                                  float* __restrict__ out) {
    int lane = threadIdx.x & 31;
    int row  = ((int)blockIdx.x * blockDim.x + (int)threadIdx.x) >> 5;
    if (row >= NNODES) return;
    int f0 = lane * 2, f1 = lane * 2 + 1;
    float2 z = ((const float2*)(in + (size_t)row * OUT_FEAT))[lane];
    float m0 = sums[f0] * (1.f / NNODES), m1 = sums[f1] * (1.f / NNODES);
    float v0 = sumsq[f0] * (1.f / NNODES) - m0 * m0;
    float v1 = sumsq[f1] * (1.f / NNODES) - m1 * m1;
    float y0 = gamma[f0] * (z.x - m0) * rsqrtf(v0 + BN_EPS) + beta[f0];
    float y1 = gamma[f1] * (z.y - m1) * rsqrtf(v1 + BN_EPS) + beta[f1];
    float mx = fmaxf(y0, y1);
#pragma unroll
    for (int off = 16; off > 0; off >>= 1) mx = fmaxf(mx, __shfl_xor(mx, off, 32));
    float s = __expf(y0 - mx) + __expf(y1 - mx);
#pragma unroll
    for (int off = 16; off > 0; off >>= 1) s += __shfl_xor(s, off, 32);
    float ls = __logf(s);
    float2 o;
    o.x = y0 - mx - ls;
    o.y = y1 - mx - ls;
    ((float2*)(out + (size_t)row * OUT_FEAT))[lane] = o;
}

// ---------------------------------------------------------------------------
extern "C" void kernel_launch(void* const* d_in, const int* in_sizes, int n_in,
                              void* d_out, int out_size, void* d_ws, size_t ws_size,
                              hipStream_t stream) {
    const float* x      = (const float*)d_in[0];
    const int*   erows  = (const int*)  d_in[1];
    const int*   ecols  = (const int*)  d_in[2];
    const float* evals  = (const float*)d_in[3];
    const float* W1     = (const float*)d_in[4];
    // d_in[5] = b1 (zeros; cancels inside BN)
    const float* gamma1 = (const float*)d_in[6];
    const float* beta1  = (const float*)d_in[7];
    const float* W2     = (const float*)d_in[8];
    // d_in[9] = b2 (zeros; cancels inside BN)
    const float* gamma2 = (const float*)d_in[10];
    const float* beta2  = (const float*)d_in[11];
    float* out = (float*)d_out;

    // Workspace carve-up (all 256B aligned; total ~128.3 MB, L2-resident on MI455X)
    char* ws = (char*)d_ws;
    size_t off = 0;
    auto carve = [&](size_t bytes) -> void* {
        void* p = ws + off;
        off += (bytes + 255) & ~(size_t)255;
        return p;
    };
    float*  xw1   = (float*) carve((size_t)NNODES * HID * 4);        // 51.2 MB
    float*  h1    = (float*) carve((size_t)NNODES * HID * 4);        // 51.2 MB
    float*  h2    = (float*) carve((size_t)NNODES * OUT_FEAT * 4);   // 12.8 MB
    float*  agg2  = (float*) carve((size_t)NNODES * OUT_FEAT * 4);   // 12.8 MB
    float*  stats = (float*) carve((size_t)(2 * HID + 2 * OUT_FEAT) * 4);
    __bf16* pW1   = (__bf16*)carve((size_t)IN_FEAT * HID * 2);       // 256 KB
    __bf16* pW2   = (__bf16*)carve((size_t)HID * OUT_FEAT * 2);      // 32 KB
    float* sums1  = stats;
    float* sumsq1 = stats + HID;
    float* sums2  = stats + 2 * HID;
    float* sumsq2 = stats + 2 * HID + OUT_FEAT;

    // 0) pack weights into bf16 WMMA fragment layout
    gcn_pack_w_bf16<<<(HID / 16) * (IN_FEAT / 32), 32, 0, stream>>>(W1, pW1, IN_FEAT, HID);
    gcn_pack_w_bf16<<<(OUT_FEAT / 16) * (HID / 32), 32, 0, stream>>>(W2, pW2, HID, OUT_FEAT);

    // 1) zero accumulation buffers + stats
    gcn_fill_zero4<<<((size_t)NNODES * HID / 4 + 255) / 256, 256, 0, stream>>>(h1, (long)NNODES * HID / 4);
    gcn_fill_zero4<<<((size_t)NNODES * OUT_FEAT / 4 + 255) / 256, 256, 0, stream>>>(agg2, (long)NNODES * OUT_FEAT / 4);
    gcn_fill_zero4<<<1, 256, 0, stream>>>(stats, (long)(2 * HID + 2 * OUT_FEAT) / 4);

    // 2) XW1 = X @ W1  (WMMA bf16, fp32 accumulate)
    gcn_gemm1<<<NNODES / 16, 128, 0, stream>>>(x, pW1, xw1);

    // 3) H1 = A @ XW1  (scatter-add, F=256)
    {
        long total = (long)NEDGES * (HID / 4);
        gcn_spmm_atomic<<<(total + 255) / 256, 256, 0, stream>>>(erows, ecols, evals, xw1, h1, HID / 4, total);
    }

    // 4) BN1 stats + fused BN+ReLU
    gcn_col_stats<<<512, HID, 0, stream>>>(h1, sums1, sumsq1, HID);
    gcn_bn_relu<<<((long)NNODES * HID + 255) / 256, 256, 0, stream>>>(h1, sums1, sumsq1, gamma1, beta1, HID, (long)NNODES * HID);

    // 5) H2 = H1 @ W2  (WMMA)
    gcn_gemm2<<<(NNODES / 16 + 3) / 4, 128, 0, stream>>>(h1, pW2, h2);

    // 6) AGG2 = A @ H2  (scatter-add, F=64)
    {
        long total = (long)NEDGES * (OUT_FEAT / 4);
        gcn_spmm_atomic<<<(total + 255) / 256, 256, 0, stream>>>(erows, ecols, evals, h2, agg2, OUT_FEAT / 4, total);
    }

    // 7) BN2 stats + fused BN+log_softmax -> out
    gcn_col_stats<<<512, OUT_FEAT, 0, stream>>>(agg2, sums2, sumsq2, OUT_FEAT);
    gcn_bn_logsoftmax<<<(NNODES * 32 + 255) / 256, 256, 0, stream>>>(agg2, sums2, sumsq2, gamma2, beta2, out);
}
